// BidirectionalLSTM_22832046145966
// MI455X (gfx1250) — matrix operand order repaired and checked by reading, not compiled, not run on hardware
//
#include <hip/hip_runtime.h>
#include <hip/hip_bf16.h>
#include <stdint.h>

// ---------------------------------------------------------------------------
// Bidirectional LSTM for MI455X (gfx1250, wave32, WMMA).
//   B=32, T=512, D=1024, H=512, 4H=2048.
// Phase 0: convert/transpose weights to bf16 [N,K]; repack X to bf16 [T*B,D]
//          (forward + length-flipped backward).
// Phase 1: Z = X @ Wi + b  via v_wmma_f32_16x16x32_bf16, output stored in
//          WMMA C-fragment tile layout (bf16) for zero-shuffle reload.
// Phase 2: persistent recurrent kernel, 32 blocks (16/direction), atomic
//          grid barrier per timestep; acc init from Z tiles, K-loop over
//          h_{t-1} @ Wh with WMMA, fp32 cell update, bf16 ping-pong h.
// ---------------------------------------------------------------------------

typedef __attribute__((ext_vector_type(16))) __bf16 v16bf;
typedef __attribute__((ext_vector_type(8)))  float  v8f;

#define DEV __device__ __forceinline__

static constexpr int Bn = 32;
static constexpr int Tn = 512;
static constexpr int Dn = 1024;
static constexpr int Hn = 512;
static constexpr int NG = 2048;   // 4*H
static constexpr int NTILES_N = NG / 16;   // 128

// ---- workspace layout (bytes, 256-aligned). total ~214 MB ----
static constexpr size_t OFF_BAR  = 0;                      // 256
static constexpr size_t OFF_HBF  = 256;                    // 2 dir * 2 buf * 32*512 * 2B
static constexpr size_t SZ_HBF   = 2ull * 2 * Bn * Hn * 2;
static constexpr size_t OFF_CST  = OFF_HBF + SZ_HBF;       // 2 dir * 32*512 * 4B
static constexpr size_t SZ_CST   = 2ull * Bn * Hn * 4;
static constexpr size_t STATE_BYTES = OFF_CST + SZ_CST;    // zeroed each launch
static constexpr size_t OFF_XF   = ((STATE_BYTES + 255) / 256) * 256;
static constexpr size_t SZ_X     = (size_t)Tn * Bn * Dn * 2;
static constexpr size_t OFF_XB   = OFF_XF + SZ_X;
static constexpr size_t OFF_WITF = OFF_XB + SZ_X;
static constexpr size_t SZ_WIT   = (size_t)NG * Dn * 2;
static constexpr size_t OFF_WITB = OFF_WITF + SZ_WIT;
static constexpr size_t OFF_WHTF = OFF_WITB + SZ_WIT;
static constexpr size_t SZ_WHT   = (size_t)NG * Hn * 2;
static constexpr size_t OFF_WHTB = OFF_WHTF + SZ_WHT;
static constexpr size_t OFF_ZF   = OFF_WHTB + SZ_WHT;
static constexpr size_t SZ_Z     = (size_t)Tn * Bn * NG * 2;
static constexpr size_t OFF_ZB   = OFF_ZF + SZ_Z;

struct Bar { unsigned count; unsigned gen; unsigned pad[14]; };

// ---- scalar bf16 helpers (RNE) ----
DEV unsigned short f2bf(float f) {
  unsigned u = __float_as_uint(f);
  u += 0x7FFFu + ((u >> 16) & 1u);
  return (unsigned short)(u >> 16);
}
DEV float bf2f(unsigned short s) { return __uint_as_float(((unsigned)s) << 16); }
DEV float sigmoidf(float x) { return 1.f / (1.f + __expf(-x)); }

// ---- WMMA fragment loaders (layouts per CDNA5 ISA §7.12.2) ----
union BFrag { uint4 u[2]; v16bf v; };

// 16-bit A 16x32: lane<16 -> row M=lane, K = {k0..k0+7, k0+16..k0+23}
//                 lane>=16 -> row M=lane-16, K = {k0+8..k0+15, k0+24..k0+31}
DEV v16bf load_a_frag(const unsigned short* __restrict__ A, int lda, int row0,
                      int k0, int lane) {
  const int l = lane & 15, hi = lane >> 4;
  const unsigned short* p = A + (size_t)(row0 + l) * lda + k0 + hi * 8;
  BFrag f;
  f.u[0] = *(const uint4*)p;          // K k0..+7 (or +8..+15)
  f.u[1] = *(const uint4*)(p + 16);   // K k0+16..+23 (or +24..+31)
  return f.v;
}

// 16-bit B 32x16 from pre-transposed weights WT[N,K]:
//   lane<16 -> N=lane, K={k0..k0+7, k0+16..k0+23} ; lane>=16 -> N=lane-16, K={k0+8..k0+15, k0+24..k0+31}
DEV v16bf load_b_frag(const unsigned short* __restrict__ WT, int ldk, int n0,
                      int k0, int lane) {
  const int l = lane & 15, hi = lane >> 4;
  const unsigned short* p = WT + (size_t)(n0 + l) * ldk + k0 + hi * 8;
  BFrag f;
  f.u[0] = *(const uint4*)p;
  f.u[1] = *(const uint4*)(p + 16);
  return f.v;
}

DEV v8f wmma_bf16(v16bf a, v16bf b, v8f c) {
  return __builtin_amdgcn_wmma_f32_16x16x32_bf16(
      /*neg_a=*/false, a, /*neg_b=*/false, b,
      /*c_mod=*/(short)0, c, /*reuse_a=*/false, /*reuse_b=*/false);
}

// ---------------------------------------------------------------------------
// Phase 0a: zero state region (barriers, h ping-pong, c)
// ---------------------------------------------------------------------------
__global__ void init_state_kernel(unsigned* __restrict__ ws_u32, size_t nwords) {
  for (size_t i = blockIdx.x * (size_t)blockDim.x + threadIdx.x; i < nwords;
       i += (size_t)gridDim.x * blockDim.x)
    ws_u32[i] = 0u;
}

// ---------------------------------------------------------------------------
// Phase 0b: W[K,N] fp32 -> WT[N,K] bf16
// ---------------------------------------------------------------------------
__global__ void wconv_kernel(const float* __restrict__ W,
                             unsigned short* __restrict__ WT, int K, int N) {
  const size_t total = (size_t)K * N;
  for (size_t i = blockIdx.x * (size_t)blockDim.x + threadIdx.x; i < total;
       i += (size_t)gridDim.x * blockDim.x) {
    const size_t n = i / K, k = i % K;
    WT[i] = f2bf(W[k * (size_t)N + n]);
  }
}

// ---------------------------------------------------------------------------
// Phase 0c: X[B,T,D] fp32 -> Xf[T*B,D] bf16 and length-flipped Xb[T*B,D] bf16
//   flipped[b,t] = x[b, (l_b-1-t) mod T]   (matches jnp roll+flip)
// ---------------------------------------------------------------------------
__global__ void xprep_kernel(const float* __restrict__ X,
                             const int* __restrict__ lengths,
                             unsigned short* __restrict__ Xf,
                             unsigned short* __restrict__ Xb) {
  const size_t total = (size_t)Tn * Bn * Dn;
  for (size_t i = blockIdx.x * (size_t)blockDim.x + threadIdx.x; i < total;
       i += (size_t)gridDim.x * blockDim.x) {
    const int d = (int)(i % Dn);
    const size_t rb = i / Dn;
    const int b = (int)(rb % Bn);
    const int t = (int)(rb / Bn);
    const int l = lengths[b];
    Xf[i] = f2bf(X[((size_t)b * Tn + t) * Dn + d]);
    int tb = l - 1 - t;
    if (tb < 0) tb += Tn;
    Xb[i] = f2bf(X[((size_t)b * Tn + tb) * Dn + d]);
  }
}

// ---------------------------------------------------------------------------
// Phase 1: Z = X @ Wi + b, stored bf16 in C-fragment tile layout.
//   grid (N/128, M/128, 2 dirs), 256 thr = 8 waves, wave tile 32Mx64N.
// ---------------------------------------------------------------------------
__global__ __launch_bounds__(256) void zgemm_kernel(
    const unsigned short* __restrict__ Xf, const unsigned short* __restrict__ Xb,
    const unsigned short* __restrict__ WiTf, const unsigned short* __restrict__ WiTb,
    const float* __restrict__ bias_f, const float* __restrict__ bias_b,
    unsigned short* __restrict__ Zf, unsigned short* __restrict__ Zb) {
  const int dir = blockIdx.z;
  const unsigned short* X  = dir ? Xb : Xf;
  const unsigned short* WT = dir ? WiTb : WiTf;
  const float* bias        = dir ? bias_b : bias_f;
  unsigned short* Z        = dir ? Zb : Zf;

  const int lane = threadIdx.x & 31;
  const int wave = threadIdx.x >> 5;
  const int wm = wave & 3;                    // 4 waves along M
  const int wn = wave >> 2;                   // 2 waves along N
  const int m0 = blockIdx.y * 128 + wm * 32;
  const int n0 = blockIdx.x * 128 + wn * 64;

  v8f acc[2][4];
#pragma unroll
  for (int i = 0; i < 2; ++i)
#pragma unroll
    for (int j = 0; j < 4; ++j) acc[i][j] = v8f{};

  for (int k0 = 0; k0 < Dn; k0 += 32) {
    const v16bf a0 = load_a_frag(X, Dn, m0, k0, lane);
    const v16bf a1 = load_a_frag(X, Dn, m0 + 16, k0, lane);
#pragma unroll
    for (int nt = 0; nt < 4; ++nt) {
      const v16bf b = load_b_frag(WT, Dn, n0 + nt * 16, k0, lane);
      acc[0][nt] = wmma_bf16(a0, b, acc[0][nt]);
      acc[1][nt] = wmma_bf16(a1, b, acc[1][nt]);
    }
  }

  // bias + fp32->bf16 + store in tile-fragment layout (512B tiles, 16B/lane)
#pragma unroll
  for (int i = 0; i < 2; ++i) {
    const int mt = (m0 >> 4) + i;
#pragma unroll
    for (int nt = 0; nt < 4; ++nt) {
      const int ntg = (n0 >> 4) + nt;
      const float bv = bias[ntg * 16 + (lane & 15)];
      unsigned pk[4];
#pragma unroll
      for (int r = 0; r < 4; ++r) {
        const unsigned lo = f2bf(acc[i][nt][2 * r] + bv);
        const unsigned hi = f2bf(acc[i][nt][2 * r + 1] + bv);
        pk[r] = lo | (hi << 16);
      }
      const size_t tidx = (size_t)mt * NTILES_N + ntg;
      *(uint4*)(Z + tidx * 256 + lane * 8) = make_uint4(pk[0], pk[1], pk[2], pk[3]);
    }
  }
}

// ---------------------------------------------------------------------------
// Phase 2: persistent recurrent kernel.
// ---------------------------------------------------------------------------
DEV void grid_barrier(Bar* bar, unsigned nb, unsigned* gen) {
  __syncthreads();
  if (threadIdx.x == 0) {
    __threadfence();
    const unsigned g = *gen;
    const unsigned arrived = atomicAdd(&bar->count, 1u);
    if (arrived == nb - 1u) {
      atomicExch(&bar->count, 0u);
      __threadfence();
      atomicAdd(&bar->gen, 1u);
    } else {
      while (atomicAdd(&bar->gen, 0u) <= g) __builtin_amdgcn_s_sleep(1);
    }
  }
  __syncthreads();
  *gen += 1;
}

__global__ __launch_bounds__(128) void lstm_rec_kernel(
    const unsigned short* __restrict__ Zf, const unsigned short* __restrict__ Zb,
    const unsigned short* __restrict__ WhTf, const unsigned short* __restrict__ WhTb,
    const int* __restrict__ lengths,
    unsigned short* __restrict__ Hbf,   // [dir][2][B*H] bf16 ping-pong
    float* __restrict__ Cst,            // [dir][B*H]
    float* __restrict__ out,            // [B,T,2H]
    float* __restrict__ fwd_final,      // [B,H]
    float* __restrict__ bwd_final,      // [B,H]
    Bar* __restrict__ bars) {
  const int lane = threadIdx.x & 31;
  const int wave = threadIdx.x >> 5;
  const int blk = blockIdx.x;             // 0..31
  const int dir = blk >> 4;               // 16 blocks / direction
  const int wg = (blk & 15) * 4 + wave;   // 0..63 waves / direction
  const int m = wg & 1;                   // batch M-tile (0..1)
  const int jg = wg >> 1;                 // 16-col group within H (0..31)

  const unsigned short* Z   = dir ? Zb : Zf;
  const unsigned short* WhT = dir ? WhTb : WhTf;
  unsigned short* Hdir = Hbf + (size_t)dir * 2 * Bn * Hn;
  float* C = Cst + (size_t)dir * Bn * Hn;
  float* fin = dir ? bwd_final : fwd_final;
  Bar* bar = bars + dir;

  const int l15 = lane & 15;
  const int rowbase = m * 16 + (lane >> 4) * 8;  // + r -> batch index
  const int ncol = jg * 16 + l15;                // 0..511 within H

  int len[8];
#pragma unroll
  for (int r = 0; r < 8; ++r) len[r] = lengths[rowbase + r];

  unsigned localGen = 0;

  for (int t = 0; t < Tn; ++t) {
    const unsigned short* Hr = Hdir + (size_t)(t & 1) * Bn * Hn;
    unsigned short* Hw = Hdir + (size_t)((t & 1) ^ 1) * Bn * Hn;

    // accumulators initialized from precomputed input projection Z (+bias)
    v8f acc[4];
#pragma unroll
    for (int g = 0; g < 4; ++g) {
      const size_t tidx = (size_t)(t * 2 + m) * NTILES_N + (g * 32 + jg);
      const uint4 zf = *(const uint4*)(Z + tidx * 256 + lane * 8);
      if (t < Tn - 1)  // prefetch next step's tile (global_prefetch_b8)
        __builtin_prefetch(Z + (tidx + 2 * NTILES_N) * 256 + lane * 8, 0, 0);
      const unsigned u[4] = {zf.x, zf.y, zf.z, zf.w};
      v8f a;
#pragma unroll
      for (int r = 0; r < 4; ++r) {
        a[2 * r]     = bf2f((unsigned short)(u[r] & 0xFFFFu));
        a[2 * r + 1] = bf2f((unsigned short)(u[r] >> 16));
      }
      acc[g] = a;
    }

    // acc += h_{t-1} @ Wh   (K = 512, four gate tiles share the A fragment)
    for (int k0 = 0; k0 < Hn; k0 += 32) {
      const v16bf a = load_a_frag(Hr, Hn, m * 16, k0, lane);
#pragma unroll
      for (int g = 0; g < 4; ++g) {
        const v16bf b = load_b_frag(WhT, Hn, g * Hn + jg * 16, k0, lane);
        acc[g] = wmma_bf16(a, b, acc[g]);
      }
    }

    // elementwise cell update; C layout: lane<16 -> M=r, lane>=16 -> M=8+r
#pragma unroll
    for (int r = 0; r < 8; ++r) {
      const int row = rowbase + r;  // batch index
      const size_t sidx = (size_t)row * Hn + ncol;
      const float iv = sigmoidf(acc[0][r]);
      const float fv = sigmoidf(acc[1][r]);
      const float gv = tanhf(acc[2][r]);
      const float ov = sigmoidf(acc[3][r]);
      const float c = fv * C[sidx] + iv * gv;
      const float h = ov * tanhf(c);
      C[sidx] = c;
      Hw[sidx] = f2bf(h);
      int tt = t;
      if (dir) { tt = len[r] - 1 - t; if (tt < 0) tt += Tn; }
      out[(size_t)row * (Tn * 2 * Hn) + (size_t)tt * (2 * Hn) + dir * Hn + ncol] = h;
      if (t == len[r] - 1) fin[(size_t)row * Hn + ncol] = h;
    }

    grid_barrier(bar, 16, &localGen);
  }
}

// ---------------------------------------------------------------------------
extern "C" void kernel_launch(void* const* d_in, const int* in_sizes, int n_in,
                              void* d_out, int out_size, void* d_ws, size_t ws_size,
                              hipStream_t stream) {
  const float* X       = (const float*)d_in[0];
  const int*   lengths = (const int*)d_in[1];
  const float* Wi_f    = (const float*)d_in[2];
  const float* Wh_f    = (const float*)d_in[3];
  const float* b_f     = (const float*)d_in[4];
  const float* Wi_b    = (const float*)d_in[5];
  const float* Wh_b    = (const float*)d_in[6];
  const float* b_b     = (const float*)d_in[7];

  char* ws = (char*)d_ws;
  Bar*            bars = (Bar*)(ws + OFF_BAR);
  unsigned short* Hbf  = (unsigned short*)(ws + OFF_HBF);
  float*          Cst  = (float*)(ws + OFF_CST);
  unsigned short* Xf   = (unsigned short*)(ws + OFF_XF);
  unsigned short* Xb   = (unsigned short*)(ws + OFF_XB);
  unsigned short* WiTf = (unsigned short*)(ws + OFF_WITF);
  unsigned short* WiTb = (unsigned short*)(ws + OFF_WITB);
  unsigned short* WhTf = (unsigned short*)(ws + OFF_WHTF);
  unsigned short* WhTb = (unsigned short*)(ws + OFF_WHTB);
  unsigned short* Zf   = (unsigned short*)(ws + OFF_ZF);
  unsigned short* Zb   = (unsigned short*)(ws + OFF_ZB);

  float* outputs   = (float*)d_out;                          // [B,T,2H]
  float* fwd_final = outputs + (size_t)Bn * Tn * 2 * Hn;     // [B,H]
  float* bwd_final = fwd_final + (size_t)Bn * Hn;            // [B,H]

  // Phase 0a: zero barriers + h ping-pong + c state
  init_state_kernel<<<260, 256, 0, stream>>>((unsigned*)ws, STATE_BYTES / 4);
  // Phase 0b: weight convert + transpose to bf16 [N,K]
  wconv_kernel<<<4096, 256, 0, stream>>>(Wi_f, WiTf, Dn, NG);
  wconv_kernel<<<4096, 256, 0, stream>>>(Wi_b, WiTb, Dn, NG);
  wconv_kernel<<<2048, 256, 0, stream>>>(Wh_f, WhTf, Hn, NG);
  wconv_kernel<<<2048, 256, 0, stream>>>(Wh_b, WhTb, Hn, NG);
  // Phase 0c: X -> bf16 [T*B,D], forward + length-flipped backward
  xprep_kernel<<<8192, 256, 0, stream>>>(X, lengths, Xf, Xb);
  // Phase 1: input-projection GEMM for both directions
  zgemm_kernel<<<dim3(NG / 128, (Tn * Bn) / 128, 2), 256, 0, stream>>>(
      Xf, Xb, WiTf, WiTb, b_f, b_b, Zf, Zb);
  // Phase 2: persistent recurrence (both directions concurrently)
  lstm_rec_kernel<<<32, 128, 0, stream>>>(Zf, Zb, WhTf, WhTb, lengths, Hbf, Cst,
                                          outputs, fwd_final, bwd_final, bars);
}
